// LinearBNNoise_31954556682658
// MI455X (gfx1250) — compile-verified, matrix operand order checked by reading, not run on hardware
//
#include <hip/hip_runtime.h>
#include <hip/hip_bf16.h>

typedef __attribute__((ext_vector_type(16))) _Float16 v16h;
typedef __attribute__((ext_vector_type(8)))  _Float16 h8;
typedef __attribute__((ext_vector_type(8)))  float    v8f;

#define EPSBN 1e-5f
#define NROWS 65536

// ---------------- deterministic noise (PCG hash + Box-Muller) ----------------
__device__ __forceinline__ unsigned pcg_hash(unsigned v) {
    v = v * 747796405u + 2891336453u;
    unsigned w = ((v >> ((v >> 28u) + 4u)) ^ v) * 277803737u;
    return (w >> 22u) ^ w;
}
__device__ __forceinline__ float nrm01(unsigned idx, unsigned seed) {
    unsigned a = pcg_hash(idx ^ seed);
    unsigned b = pcg_hash(a + 0x9E3779B9u);
    float u1 = (float)(a >> 8) * (1.0f / 16777216.0f) + 1.0e-7f;
    float u2 = (float)(b >> 8) * (1.0f / 16777216.0f);
    float r  = sqrtf(-2.0f * __logf(u1));
    return r * __cosf(6.28318530718f * u2);
}

// ---------------- WMMA fragment load from LDS (16-bit A/B layout) ------------
__device__ __forceinline__ v16h frag_ld(const _Float16* lds, int row0, int stride, int kbase) {
    const int lane = threadIdx.x & 31;
    const int r  = row0 + (lane & 15);
    const int kb = kbase + ((lane & 16) ? 8 : 0);
    const _Float16* p = lds + r * stride + kb;
    union { v16h v; _Float16 h[16]; } u;
#pragma unroll
    for (int i = 0; i < 8; ++i) { u.h[i] = p[i]; u.h[i + 8] = p[i + 16]; }
    return u.v;
}

// ---------------- zero stats region ------------------------------------------
__global__ void zero_stats_kernel(float* ws) {
    int t = threadIdx.x;
    if (t < 512) ws[t] = 0.0f;
}

// ---------------- GEMM1 tile staging helpers ---------------------------------
__device__ __forceinline__ void load_tiles(const float* __restrict__ x,
                                           const float* __restrict__ w1,
                                           long rowBase, int kb, int kw, int tid,
                                           float4 rx[4], float4 rw[2]) {
#pragma unroll
    for (int i = 0; i < 4; ++i) {                 // X tile: 128 rows x 32 cols
        int id = tid + i * 256;                   // 0..1023
        int r = id >> 3, c4 = (id & 7) * 4;
        if (c4 < kw) rx[i] = *(const float4*)(x + (rowBase + r) * 784 + kb + c4);
        else         rx[i] = make_float4(0.f, 0.f, 0.f, 0.f);
    }
#pragma unroll
    for (int i = 0; i < 2; ++i) {                 // W tile: 48 rows x 32 cols
        int id = tid + i * 256;
        if (id < 384) {
            int r = id >> 3, c4 = (id & 7) * 4;
            if (c4 < kw) rw[i] = *(const float4*)(w1 + r * 784 + kb + c4);
            else         rw[i] = make_float4(0.f, 0.f, 0.f, 0.f);
        }
    }
}
__device__ __forceinline__ void store_tiles(_Float16* sX, _Float16* sW, int tid,
                                            const float4 rx[4], const float4 rw[2]) {
#pragma unroll
    for (int i = 0; i < 4; ++i) {
        int id = tid + i * 256;
        int r = id >> 3, c4 = (id & 7) * 4;
        _Float16* d = &sX[r * 40 + c4];
        d[0] = (_Float16)rx[i].x; d[1] = (_Float16)rx[i].y;
        d[2] = (_Float16)rx[i].z; d[3] = (_Float16)rx[i].w;
    }
#pragma unroll
    for (int i = 0; i < 2; ++i) {
        int id = tid + i * 256;
        if (id < 384) {
            int r = id >> 3, c4 = (id & 7) * 4;
            _Float16* d = &sW[r * 40 + c4];
            d[0] = (_Float16)rw[i].x; d[1] = (_Float16)rw[i].y;
            d[2] = (_Float16)rw[i].z; d[3] = (_Float16)rw[i].w;
        }
    }
}

// ---------------- GEMM1: y1 = x(65536x784) * w1^T + b1 (f16 out), + stats ----
__global__ __launch_bounds__(256) void gemm1_kernel(const float* __restrict__ x,
                                                    const float* __restrict__ w1,
                                                    const float* __restrict__ b1,
                                                    _Float16* __restrict__ y1h,
                                                    float* __restrict__ gsum,
                                                    float* __restrict__ gsq) {
    __shared__ _Float16 sX[2][128 * 40];
    __shared__ _Float16 sW[2][48 * 40];
    __shared__ float red[96];
    const int tid  = threadIdx.x;
    const int wid  = tid >> 5;
    const int lane = tid & 31;
    const long rowBase = (long)blockIdx.x * 128;

    v8f acc0 = {}, acc1 = {}, acc2 = {};
    float4 rx[4]; float4 rw[2];

    // prologue: stage tile 0
    load_tiles(x, w1, rowBase, 0, 32, tid, rx, rw);
    store_tiles(sX[0], sW[0], tid, rx, rw);
    __syncthreads();

    for (int ks = 0; ks < 25; ++ks) {
        const int cur = ks & 1;
        const bool more = (ks + 1 < 25);
        if (more) {  // issue next tile's global loads early (overlap with WMMA)
            const int kb2 = (ks + 1) * 32;
            const int kw2 = (ks + 1 == 24) ? 16 : 32;
            load_tiles(x, w1, rowBase, kb2, kw2, tid, rx, rw);
        }
        v16h a   = frag_ld(sX[cur], wid * 16, 40, 0);
        v16h b0f = frag_ld(sW[cur],  0, 40, 0);
        v16h b1f = frag_ld(sW[cur], 16, 40, 0);
        v16h b2f = frag_ld(sW[cur], 32, 40, 0);
        acc0 = __builtin_amdgcn_wmma_f32_16x16x32_f16(false, a, false, b0f, (short)0, acc0, false, false);
        acc1 = __builtin_amdgcn_wmma_f32_16x16x32_f16(false, a, false, b1f, (short)0, acc1, false, false);
        acc2 = __builtin_amdgcn_wmma_f32_16x16x32_f16(false, a, false, b2f, (short)0, acc2, false, false);
        if (more) store_tiles(sX[cur ^ 1], sW[cur ^ 1], tid, rx, rw);
        __syncthreads();
    }

    const int nloc = lane & 15;
    const int rOff = (lane & 16) ? 8 : 0;
    const float bv0 = b1[nloc], bv1 = b1[16 + nloc], bv2 = b1[32 + nloc];

    if (tid < 96) red[tid] = 0.0f;
    __syncthreads();

#pragma unroll
    for (int t = 0; t < 3; ++t) {
        v8f acc = (t == 0) ? acc0 : ((t == 1) ? acc1 : acc2);
        const float bv = (t == 0) ? bv0 : ((t == 1) ? bv1 : bv2);
        const int n = t * 16 + nloc;
        float s = 0.0f, q = 0.0f;
#pragma unroll
        for (int e = 0; e < 8; ++e) {
            float v = acc[e] + bv;
            s += v; q += v * v;
            y1h[(rowBase + wid * 16 + rOff + e) * 48 + n] = (_Float16)v;
        }
        atomicAdd(&red[n], s);
        atomicAdd(&red[48 + n], q);
    }
    __syncthreads();
    if (tid < 48) {
        atomicAdd(&gsum[tid], red[tid]);
        atomicAdd(&gsq[tid],  red[48 + tid]);
    }
}

// ---------------- BN + noise stats -> folded (scale, shift), noise std -------
__global__ void bnstats_kernel(const float* __restrict__ gsum, const float* __restrict__ gsq,
                               const float* __restrict__ gamma, const float* __restrict__ beta,
                               float* __restrict__ sc_out, float* __restrict__ sh_out,
                               float* __restrict__ scal_out, int nfeat, float ufac) {
    __shared__ float hs[64], hq[64];
    const int t = threadIdx.x;  // 64 threads
    float s = 0.0f, q = 0.0f, mean = 0.0f, sc = 0.0f;
    if (t < nfeat) {
        mean = gsum[t] * (1.0f / (float)NROWS);
        float var = fmaxf(gsq[t] * (1.0f / (float)NROWS) - mean * mean, 0.0f);
        float r = rsqrtf(var + EPSBN);
        sc = gamma[t] * r;
        s = (float)NROWS * beta[t];
        q = (float)NROWS * (var / (var + EPSBN) * gamma[t] * gamma[t] + beta[t] * beta[t]);
    }
    hs[t] = s; hq[t] = q;
    __syncthreads();
    for (int st = 32; st > 0; st >>= 1) {
        if (t < st) { hs[t] += hs[t + st]; hq[t] += hq[t + st]; }
        __syncthreads();
    }
    float N = (float)NROWS * (float)nfeat;
    float nmean = hs[0] / N;
    if (t == 0) {
        float var = (hq[0] - N * nmean * nmean) / (N - 1.0f);
        scal_out[0] = nmean;
        scal_out[1] = sqrtf(fmaxf(var, 0.0f)) * ufac;
    }
    if (t < nfeat) {
        sc_out[t] = sc;
        sh_out[t] = beta[t] - mean * sc + nmean;   // fold noise mean into shift
    }
}

// ---------------- GEMM2: h1=relu(y1*sc+sh+noise); y2 = h1*w2^T + b2, stats ---
__global__ __launch_bounds__(256) void gemm2_kernel(const _Float16* __restrict__ y1h,
                                                    const float* __restrict__ sc1,
                                                    const float* __restrict__ sh1,
                                                    const float* __restrict__ w2,
                                                    const float* __restrict__ b2,
                                                    const float* __restrict__ scal,
                                                    _Float16* __restrict__ y2h,
                                                    float* __restrict__ gsum2,
                                                    float* __restrict__ gsq2) {
    __shared__ _Float16 sH[128 * 72];   // K padded to 64 (zeros in 48..63)
    __shared__ _Float16 sW[32 * 72];    // N padded to 32, K padded to 64
    __shared__ float red[48];
    __shared__ float sS[48], sHf[48];
    const int tid = threadIdx.x, wid = tid >> 5, lane = tid & 31;
    const long rowBase = (long)blockIdx.x * 128;
    const float ns = scal[1];

    if (tid < 48) { sS[tid] = sc1[tid]; sHf[tid] = sh1[tid]; }
    for (int i = tid; i < 2048; i += 256) {                 // zero pad cols 48..63
        int r = i >> 4, c = 48 + (i & 15);
        sH[r * 72 + c] = (_Float16)0.0f;
    }
    for (int i = tid; i < 32 * 72; i += 256) sW[i] = (_Float16)0.0f;
    __syncthreads();
    for (int i = tid; i < 24 * 48; i += 256) {              // w2: 24x48
        int n = i / 48, k = i % 48;
        sW[n * 72 + k] = (_Float16)w2[i];
    }
#pragma unroll
    for (int i = 0; i < 3; ++i) {                           // h tile: 128 x 48 (f16 in)
        int id = tid + i * 256;                             // 0..767 h8 chunks
        int r = id / 6, c8 = (id % 6) * 8;
        long gi = (rowBase + r) * 48 + c8;
        h8 f = *(const h8*)(y1h + gi);
#pragma unroll
        for (int j = 0; j < 8; ++j) {
            int c = c8 + j;
            float v = (float)f[j] * sS[c] + sHf[c] + ns * nrm01((unsigned)(gi + j), 0x1234u);
            sH[r * 72 + c] = (_Float16)fmaxf(v, 0.0f);
        }
    }
    __syncthreads();

    v8f acc0 = {}, acc1 = {};
    v16h a0  = frag_ld(sH, wid * 16, 72, 0);
    v16h a1  = frag_ld(sH, wid * 16, 72, 32);
    v16h b00 = frag_ld(sW,  0, 72, 0);
    v16h b01 = frag_ld(sW,  0, 72, 32);
    v16h b10 = frag_ld(sW, 16, 72, 0);
    v16h b11 = frag_ld(sW, 16, 72, 32);
    acc0 = __builtin_amdgcn_wmma_f32_16x16x32_f16(false, a0, false, b00, (short)0, acc0, false, false);
    acc0 = __builtin_amdgcn_wmma_f32_16x16x32_f16(false, a1, false, b01, (short)0, acc0, false, false);
    acc1 = __builtin_amdgcn_wmma_f32_16x16x32_f16(false, a0, false, b10, (short)0, acc1, false, false);
    acc1 = __builtin_amdgcn_wmma_f32_16x16x32_f16(false, a1, false, b11, (short)0, acc1, false, false);

    const int nloc = lane & 15;
    const int rOff = (lane & 16) ? 8 : 0;
    __syncthreads();
    if (tid < 48) red[tid] = 0.0f;
    __syncthreads();
    { // tile 0: n = 0..15 (all valid)
        const float bv = b2[nloc];
        float s = 0.0f, q = 0.0f;
#pragma unroll
        for (int e = 0; e < 8; ++e) {
            float v = acc0[e] + bv;
            s += v; q += v * v;
            y2h[(rowBase + wid * 16 + rOff + e) * 24 + nloc] = (_Float16)v;
        }
        atomicAdd(&red[nloc], s);
        atomicAdd(&red[24 + nloc], q);
    }
    if (nloc < 8) { // tile 1: n = 16..23 valid
        const int n = 16 + nloc;
        const float bv = b2[n];
        float s = 0.0f, q = 0.0f;
#pragma unroll
        for (int e = 0; e < 8; ++e) {
            float v = acc1[e] + bv;
            s += v; q += v * v;
            y2h[(rowBase + wid * 16 + rOff + e) * 24 + n] = (_Float16)v;
        }
        atomicAdd(&red[n], s);
        atomicAdd(&red[24 + n], q);
    }
    __syncthreads();
    if (tid < 24) {
        atomicAdd(&gsum2[tid], red[tid]);
        atomicAdd(&gsq2[tid],  red[24 + tid]);
    }
}

// ---------------- GEMM3: h2=relu(y2*sc+sh+noise); out = h2*w3^T + b3 ---------
__global__ __launch_bounds__(256) void gemm3_kernel(const _Float16* __restrict__ y2h,
                                                    const float* __restrict__ sc2,
                                                    const float* __restrict__ sh2,
                                                    const float* __restrict__ w3,
                                                    const float* __restrict__ b3,
                                                    const float* __restrict__ scal,
                                                    float* __restrict__ out) {
    __shared__ _Float16 sH[128 * 40];   // K=24 padded to 32
    __shared__ _Float16 sW[16 * 40];    // N=10 padded to 16
    __shared__ float sS[24], sHf[24];
    const int tid = threadIdx.x, wid = tid >> 5, lane = tid & 31;
    const long rowBase = (long)blockIdx.x * 128;
    const float ns = scal[1];

    if (tid < 24) { sS[tid] = sc2[tid]; sHf[tid] = sh2[tid]; }
    for (int i = tid; i < 1024; i += 256) {                 // zero pad cols 24..31
        int r = i >> 3, c = 24 + (i & 7);
        sH[r * 40 + c] = (_Float16)0.0f;
    }
    for (int i = tid; i < 16 * 40; i += 256) sW[i] = (_Float16)0.0f;
    __syncthreads();
    for (int i = tid; i < 10 * 24; i += 256) {              // w3: 10x24
        int n = i / 24, k = i % 24;
        sW[n * 40 + k] = (_Float16)w3[i];
    }
#pragma unroll
    for (int i = 0; i < 2; ++i) {                           // h tile: 128 x 24 (f16 in)
        int id = tid + i * 256;                             // 0..383 h8 chunks
        if (id < 384) {
            int r = id / 3, c8 = (id % 3) * 8;
            long gi = (rowBase + r) * 24 + c8;
            h8 f = *(const h8*)(y2h + gi);
#pragma unroll
            for (int j = 0; j < 8; ++j) {
                int c = c8 + j;
                float v = (float)f[j] * sS[c] + sHf[c] + ns * nrm01((unsigned)(gi + j), 0x5678u);
                sH[r * 40 + c] = (_Float16)fmaxf(v, 0.0f);
            }
        }
    }
    __syncthreads();

    v8f acc = {};
    v16h a = frag_ld(sH, wid * 16, 40, 0);
    v16h b = frag_ld(sW, 0, 40, 0);
    acc = __builtin_amdgcn_wmma_f32_16x16x32_f16(false, a, false, b, (short)0, acc, false, false);

    const int nloc = lane & 15;
    const int rOff = (lane & 16) ? 8 : 0;
    if (nloc < 10) {
        const float bv = b3[nloc];
#pragma unroll
        for (int e = 0; e < 8; ++e) {
            out[(rowBase + wid * 16 + rOff + e) * 10 + nloc] = acc[e] + bv;
        }
    }
}

// ---------------- launcher ----------------------------------------------------
extern "C" void kernel_launch(void* const* d_in, const int* in_sizes, int n_in,
                              void* d_out, int out_size, void* d_ws, size_t ws_size,
                              hipStream_t stream) {
    const float* x   = (const float*)d_in[0];
    const float* w1  = (const float*)d_in[1];
    const float* b1  = (const float*)d_in[2];
    const float* g1  = (const float*)d_in[3];
    const float* be1 = (const float*)d_in[4];
    const float* w2  = (const float*)d_in[5];
    const float* b2  = (const float*)d_in[6];
    const float* g2  = (const float*)d_in[7];
    const float* be2 = (const float*)d_in[8];
    const float* w3  = (const float*)d_in[9];
    const float* b3  = (const float*)d_in[10];
    float* out = (float*)d_out;
    float* ws  = (float*)d_ws;

    // workspace layout (floats)
    float* sum1  = ws + 0;     // 48
    float* sq1   = ws + 48;    // 48
    float* sc1   = ws + 96;    // 48
    float* sh1   = ws + 144;   // 48
    float* sum2  = ws + 192;   // 24
    float* sq2   = ws + 216;   // 24
    float* sc2   = ws + 240;   // 24
    float* sh2   = ws + 264;   // 24
    float* scal1 = ws + 288;   // 2
    float* scal2 = ws + 290;   // 2
    _Float16* Y1h = (_Float16*)(ws + 512);                  // 65536*48 halves
    _Float16* Y2h = (_Float16*)(ws + 512 + 1572864);        // 65536*24 halves

    zero_stats_kernel<<<1, 256, 0, stream>>>(ws);
    gemm1_kernel<<<512, 256, 0, stream>>>(x, w1, b1, Y1h, sum1, sq1);
    bnstats_kernel<<<1, 64, 0, stream>>>(sum1, sq1, g1, be1, sc1, sh1, scal1, 48, 1.4711f);
    gemm2_kernel<<<512, 256, 0, stream>>>(Y1h, sc1, sh1, w2, b2, scal1, Y2h, sum2, sq2);
    bnstats_kernel<<<1, 64, 0, stream>>>(sum2, sq2, g2, be2, sc2, sh2, scal2, 24, 1.8223f);
    gemm3_kernel<<<512, 256, 0, stream>>>(Y2h, sc2, sh2, w3, b3, scal2, out);
}
